// AttentionCircuit_52450140618980
// MI455X (gfx1250) — matrix operand order repaired and checked by reading, not compiled, not run on hardware
//
#include <hip/hip_runtime.h>

// ---------------------------------------------------------------------------
// AttentionCircuit for MI455X (gfx1250, wave32, WMMA bf16 -> f32 accumulate)
//
// Pipeline (all GEMMs via v_wmma_f32_16x16x32_bf16, tiles staged via TDM):
//   1. convert x, f_qk, f_v (permuted), r_qk, r_v, W_O^T to bf16 scratch
//   2. T_qk = X @ Fqk (4096x1024 @ 1024x2048), T_v = X @ Fv      [2 GEMMs]
//   3. h_{q,k,v}[s,r] = sum_n w[s,n] * T[s, n*128+r]             [reduce]
//   4. WH[s, n*128+r] = w[s,n] * h[s,r]                          [outer]
//   5. Q/K/V = WH @ r_flat (4096x2048 @ 2048x1024) -> bf16       [3 GEMMs]
//   6. causal flash attention, 16 heads x 64 dh, online softmax
//   7. out = attn @ W_O^T (4096x1024 @ 1024x1024) -> f32 d_out
// ---------------------------------------------------------------------------

typedef __bf16 bf16;
typedef __attribute__((ext_vector_type(16))) __bf16 bf16x16;
typedef __attribute__((ext_vector_type(8)))  float  v8f;
typedef __attribute__((ext_vector_type(4)))  unsigned int u32x4;
typedef __attribute__((ext_vector_type(8)))  int i32x8;
typedef __attribute__((ext_vector_type(4)))  int i32x4;

union Frag { u32x4 u[2]; bf16x16 v; };

__device__ __forceinline__ bf16 f2bf(float f) {
    union { float f; unsigned u; } a; a.f = f;
    unsigned r = a.u + 0x7FFFu + ((a.u >> 16) & 1u);   // round-to-nearest-even
    union { unsigned short s; bf16 b; } c;
    c.s = (unsigned short)(r >> 16);
    return c.b;
}

// 2D bf16 tile load global->LDS via Tensor Data Mover.
// lds_off: LDS byte offset; rows of tile_d0 elems, row stride stride_elems;
// pad_int_enc: encoded pad interval (ISA: 0=2,1=4,2=8,3=16,4=32,... DWORDs);
// pad_dw: padding amount in DWORDs (0 = no padding).
__device__ __forceinline__ void tdm_load_2d(unsigned lds_off, const void* gptr,
                                            unsigned tens_d0, unsigned tens_d1,
                                            unsigned stride_elems,
                                            unsigned tile_d0, unsigned tile_d1,
                                            unsigned pad_int_enc, unsigned pad_dw) {
    unsigned long long ga = (unsigned long long)(size_t)gptr;
    u32x4 g0;
    g0[0] = 1u;                                        // count=1 user D#
    g0[1] = lds_off;                                   // lds_addr
    g0[2] = (unsigned)(ga & 0xffffffffu);              // global_addr[31:0]
    g0[3] = (unsigned)((ga >> 32) & 0x01ffffffu) | (2u << 30); // addr[56:32], type=2
    i32x8 g1;
    unsigned pe = pad_dw ? 1u : 0u;
    unsigned pa = pad_dw ? (pad_dw - 1u) : 0u;
    g1[0] = (int)((1u << 16) | (pe << 20) | (pad_int_enc << 22) | (pa << 25)); // data_size=2B
    g1[1] = (int)((tens_d0 & 0xffffu) << 16);          // abar=0 | tensor_dim0 lo
    g1[2] = (int)((tens_d0 >> 16) | ((tens_d1 & 0xffffu) << 16));
    g1[3] = (int)((tens_d1 >> 16) | (tile_d0 << 16));  // tensor_dim1 hi | tile_dim0
    g1[4] = (int)tile_d1;                              // tile_dim1 | tile_dim2=0
    g1[5] = (int)stride_elems;                         // tensor_dim0_stride[31:0]
    g1[6] = 0;                                         // stride hi | dim1_stride lo
    g1[7] = 0;
    i32x4 z4 = {0, 0, 0, 0};
    i32x8 z8 = {0, 0, 0, 0, 0, 0, 0, 0};
    __builtin_amdgcn_tensor_load_to_lds(g0, g1, z4, z4, z8, 0);
}

#define CB  2
#define CS  2048
#define CD  1024
#define CR  128
#define CH  16
#define CN  16
#define CBS 4096
#define CNR 2048
#define CDH 64

// ------------------------- conversion kernels ------------------------------

__global__ __launch_bounds__(256) void k_cvt(const float* __restrict__ s,
                                             bf16* __restrict__ d, int n) {
    int i = blockIdx.x * 256 + threadIdx.x;
    if (i < n) d[i] = f2bf(s[i]);
}

// f (N,D,R) -> Fflat (D, N*R):  Fflat[d, n*R+r] = f[n,d,r]
__global__ __launch_bounds__(256) void k_cvt_feat(const float* __restrict__ f,
                                                  bf16* __restrict__ d) {
    int idx = blockIdx.x * 256 + threadIdx.x;          // over N*D*R
    int r = idx & (CR - 1);
    int dd = (idx >> 7) & (CD - 1);
    int n = idx >> 17;
    d[(size_t)dd * CNR + n * CR + r] = f2bf(f[idx]);
}

// W_O (D,D) -> WOt (K=e, N=d):  WOt[e*D + d] = W_O[d*D + e]
__global__ __launch_bounds__(256) void k_cvt_wot(const float* __restrict__ w,
                                                 bf16* __restrict__ d) {
    int idx = blockIdx.x * 256 + threadIdx.x;          // over D*D
    int dd = idx >> 10;
    int e  = idx & (CD - 1);
    d[(size_t)e * CD + dd] = f2bf(w[idx]);
}

// h[s,r] = sum_n w[s,n] * T[s, n*R + r]
__global__ __launch_bounds__(256) void k_reduce_h(const float* __restrict__ T,
                                                  const float* __restrict__ w,
                                                  float* __restrict__ h) {
    int idx = blockIdx.x * 256 + threadIdx.x;          // over BS*R
    int s = idx >> 7;
    int r = idx & (CR - 1);
    const float* trow = T + (size_t)s * CNR + r;
    const float* wrow = w + s * CN;
    float acc = 0.f;
#pragma unroll
    for (int n = 0; n < CN; n++) acc += wrow[n] * trow[n * CR];
    h[idx] = acc;
}

// WH[s, n*R+r] = w[s,n] * h[s,r]  (bf16)
__global__ __launch_bounds__(256) void k_build_wh(const float* __restrict__ h,
                                                  const float* __restrict__ w,
                                                  bf16* __restrict__ wh) {
    int idx = blockIdx.x * 256 + threadIdx.x;          // over BS*NR
    int s = idx >> 11;
    int nr = idx & (CNR - 1);
    int n = nr >> 7;
    int r = nr & (CR - 1);
    wh[idx] = f2bf(w[s * CN + n] * h[s * CR + r]);
}

// ------------------------------ tiled GEMM ---------------------------------
// C(MxN) = A(MxK, bf16 row-major) @ B(KxN, bf16 row-major); f32 or bf16 out.
// 128x128 block, BK=32, 8 waves: each wave 32x64 = 2x4 WMMA tiles.
// A tile staged by TDM (padded rows), B tile transposed manually.

#define BM 128
#define BN 128
#define BK 32
#define LDP 40   // padded LDS leading dim (bf16): 80B = odd multiple of 16B

template <bool OUT_BF16>
__global__ __launch_bounds__(256) void k_gemm(const bf16* __restrict__ A,
                                              const bf16* __restrict__ B,
                                              void* __restrict__ Cout,
                                              int M, int N, int K) {
    __shared__ __align__(16) bf16 sA[BM * LDP];
    __shared__ __align__(16) bf16 sB[BN * LDP];   // transposed: [n][k]

    const int tid  = threadIdx.x;
    const int lane = tid & 31;
    const int w    = tid >> 5;
    const int wm   = (w & 3) * 32;
    const int wn   = (w >> 2) * 64;
    const int m0   = blockIdx.y * BM;
    const int n0   = blockIdx.x * BN;
    const int hl   = lane >> 4;       // half-wave
    const int l15  = lane & 15;
    const unsigned sA_off = (unsigned)(size_t)sA;

    v8f acc[2][4];
#pragma unroll
    for (int i = 0; i < 2; i++)
#pragma unroll
        for (int j = 0; j < 4; j++) acc[i][j] = {};

    for (int k0 = 0; k0 < K; k0 += BK) {
        // stage A tile 128x32 via TDM: rows 64B, pad 4 DWORDs every 16 DWORDs
        if (w == 0)
            tdm_load_2d(sA_off, A + (size_t)m0 * K + k0,
                        (unsigned)K, (unsigned)M, (unsigned)K,
                        BK, BM, /*interval 16DW*/3, /*pad*/4);
        // stage B tile 32x128 transposed into sB[n][k]
        for (int i = tid; i < BK * (BN / 2); i += 256) {
            int k  = i >> 6;
            int np = (i & 63) * 2;
            const bf16* src = B + (size_t)(k0 + k) * N + n0 + np;
            sB[np * LDP + k]       = src[0];
            sB[(np + 1) * LDP + k] = src[1];
        }
        if (k0 + BK < K)   // global_prefetch_b8 of next B tile
            __builtin_prefetch(B + (size_t)(k0 + BK + (tid >> 3)) * N + n0, 0, 1);
        if (w == 0) __builtin_amdgcn_s_wait_tensorcnt(0);
        __syncthreads();

        Frag a[2];
#pragma unroll
        for (int i = 0; i < 2; i++) {
            const bf16* p = sA + (wm + i * 16 + l15) * LDP;
            a[i].u[0] = *(const u32x4*)(p + 8 * hl);
            a[i].u[1] = *(const u32x4*)(p + 16 + 8 * hl);
        }
        Frag b[4];
#pragma unroll
        for (int j = 0; j < 4; j++) {
            const bf16* p = sB + (wn + j * 16 + l15) * LDP + 8 * hl;
            b[j].u[0] = *(const u32x4*)(p);
            b[j].u[1] = *(const u32x4*)(p + 16);
        }
#pragma unroll
        for (int i = 0; i < 2; i++)
#pragma unroll
            for (int j = 0; j < 4; j++)
                acc[i][j] = __builtin_amdgcn_wmma_f32_16x16x32_bf16(
                    false, a[i].v, false, b[j].v, (short)0, acc[i][j], false, false);
        __syncthreads();
    }

#pragma unroll
    for (int i = 0; i < 2; i++)
#pragma unroll
        for (int j = 0; j < 4; j++) {
            int row0 = m0 + wm + i * 16;
            int col  = n0 + wn + j * 16 + l15;
            if (OUT_BF16) {
                bf16* C = (bf16*)Cout;
#pragma unroll
                for (int v = 0; v < 8; v++)
                    C[(size_t)(row0 + v + 8 * hl) * N + col] = f2bf(acc[i][j][v]);
            } else {
                float* C = (float*)Cout;
#pragma unroll
                for (int v = 0; v < 8; v++)
                    C[(size_t)(row0 + v + 8 * hl) * N + col] = acc[i][j][v];
            }
        }
}

// --------------------------- flash attention -------------------------------
// grid (S/128, B*H); 8 waves/WG, each wave owns 16 query rows; key blocks of 32.
// K tile staged by TDM (padded rows), V tile transposed manually.

__global__ __launch_bounds__(256) void k_attn(const bf16* __restrict__ Q,
                                              const bf16* __restrict__ Kb,
                                              const bf16* __restrict__ V,
                                              bf16* __restrict__ O) {
    __shared__ __align__(16) bf16 sK[32 * 72];    // [key][dh], padded
    __shared__ __align__(16) bf16 sVt[64 * 40];   // [dh][key], padded
    __shared__ __align__(16) bf16 sP[8 * 16 * 40];// per-wave P tile, padded

    const int tid  = threadIdx.x;
    const int lane = tid & 31;
    const int w    = tid >> 5;
    const int hl   = lane >> 4;
    const int l15  = lane & 15;
    const int qb   = blockIdx.x;
    const int bh   = blockIdx.y;
    const int bb   = bh >> 4;
    const int hh   = bh & 15;
    const size_t base = (size_t)bb * CS * CD + hh * CDH;  // + s*CD
    const int qrow0 = qb * 128 + w * 16;
    const unsigned sK_off = (unsigned)(size_t)sK;

    Frag qf[2];
    {
        const bf16* p = Q + base + (size_t)(qrow0 + l15) * CD;
#pragma unroll
        for (int i = 0; i < 2; i++) {
            qf[i].u[0] = *(const u32x4*)(p + i * 32 + 8 * hl);
            qf[i].u[1] = *(const u32x4*)(p + i * 32 + 16 + 8 * hl);
        }
    }

    v8f o[4];
#pragma unroll
    for (int j = 0; j < 4; j++) o[j] = {};
    float mrow[8], lrow[8];
#pragma unroll
    for (int v = 0; v < 8; v++) { mrow[v] = -__builtin_inff(); lrow[v] = 0.f; }

    const int nkb = (qb + 1) * 4;
    for (int kb = 0; kb < nkb; kb++) {
        const int kst = kb * 32;
        // stage K block 32x64 via TDM: rows 128B, pad 4 DWORDs every 32 DWORDs
        if (w == 0)
            tdm_load_2d(sK_off, Kb + base + (size_t)kst * CD,
                        CD, CBS, CD, CDH, 32, /*interval 32DW*/4, /*pad*/4);
        for (int i = tid; i < 2048; i += 256) {   // stage V transposed 64x32
            int key = i >> 6, dh = i & 63;
            sVt[dh * 40 + key] = V[base + (size_t)(kst + key) * CD + dh];
        }
        if (w == 0) __builtin_amdgcn_s_wait_tensorcnt(0);
        __syncthreads();

        if (kst <= qrow0 + 15) {                  // wave-uniform causal skip
            v8f sc[2];
#pragma unroll
            for (int nt = 0; nt < 2; nt++) {
                sc[nt] = {};
#pragma unroll
                for (int ks = 0; ks < 2; ks++) {
                    Frag kf;
                    const bf16* p = sK + (nt * 16 + l15) * 72 + ks * 32 + 8 * hl;
                    kf.u[0] = *(const u32x4*)(p);
                    kf.u[1] = *(const u32x4*)(p + 16);
                    sc[nt] = __builtin_amdgcn_wmma_f32_16x16x32_bf16(
                        false, qf[ks].v, false, kf.v, (short)0, sc[nt], false, false);
                }
            }
            float alpha[8];
#pragma unroll
            for (int v = 0; v < 8; v++) {
                const int row = qrow0 + v + 8 * hl;
#pragma unroll
                for (int nt = 0; nt < 2; nt++) {
                    float x = sc[nt][v] * 0.125f;
                    int col = kst + nt * 16 + l15;
                    sc[nt][v] = (col <= row) ? x : -__builtin_inff();
                }
                float t = fmaxf(sc[0][v], sc[1][v]);
#pragma unroll
                for (int m = 1; m < 16; m <<= 1) t = fmaxf(t, __shfl_xor(t, m, 32));
                float mn = fmaxf(mrow[v], t);
                float al = __expf(mrow[v] - mn);
                float p0 = __expf(sc[0][v] - mn);
                float p1 = __expf(sc[1][v] - mn);
                float rs = p0 + p1;
#pragma unroll
                for (int m = 1; m < 16; m <<= 1) rs += __shfl_xor(rs, m, 32);
                lrow[v] = lrow[v] * al + rs;
                mrow[v] = mn;
                alpha[v] = al;
                bf16* sp = sP + w * 640 + (v + 8 * hl) * 40;
                sp[l15]      = f2bf(p0);
                sp[16 + l15] = f2bf(p1);
            }
#pragma unroll
            for (int j = 0; j < 4; j++)
#pragma unroll
                for (int v = 0; v < 8; v++) o[j][v] *= alpha[v];

            Frag pf;   // LDS ops are in-order within a wave
            {
                const bf16* p = sP + w * 640 + l15 * 40;
                pf.u[0] = *(const u32x4*)(p + 8 * hl);
                pf.u[1] = *(const u32x4*)(p + 16 + 8 * hl);
            }
#pragma unroll
            for (int j = 0; j < 4; j++) {
                Frag vf;
                const bf16* p = sVt + (j * 16 + l15) * 40 + 8 * hl;
                vf.u[0] = *(const u32x4*)(p);
                vf.u[1] = *(const u32x4*)(p + 16);
                o[j] = __builtin_amdgcn_wmma_f32_16x16x32_bf16(
                    false, pf.v, false, vf.v, (short)0, o[j], false, false);
            }
        }
        __syncthreads();
    }

#pragma unroll
    for (int v = 0; v < 8; v++) {
        float inv = 1.f / lrow[v];
        int row = qrow0 + v + 8 * hl;
#pragma unroll
        for (int j = 0; j < 4; j++)
            O[base + (size_t)row * CD + j * 16 + l15] = f2bf(o[j][v] * inv);
    }
}

// ------------------------------- launcher ----------------------------------

extern "C" void kernel_launch(void* const* d_in, const int* in_sizes, int n_in,
                              void* d_out, int out_size, void* d_ws, size_t ws_size,
                              hipStream_t stream) {
    const float* x   = (const float*)d_in[0];
    const float* wfQ = (const float*)d_in[1];
    const float* wfK = (const float*)d_in[2];
    const float* wfV = (const float*)d_in[3];
    const float* wrQ = (const float*)d_in[4];
    const float* wrK = (const float*)d_in[5];
    const float* wrV = (const float*)d_in[6];
    const float* fqk = (const float*)d_in[7];
    const float* fv  = (const float*)d_in[8];
    const float* rqk = (const float*)d_in[9];
    const float* rv  = (const float*)d_in[10];
    const float* wo  = (const float*)d_in[11];

    char* ws = (char*)d_ws;
    size_t off = 0;
    auto bump = [&](size_t bytes) { size_t r = off; off += (bytes + 255) & ~(size_t)255; return r; };

    bf16* Xb   = (bf16*)(ws + bump((size_t)CBS * CD * 2));
    bf16* Fqk  = (bf16*)(ws + bump((size_t)CD * CNR * 2));
    bf16* Fv   = (bf16*)(ws + bump((size_t)CD * CNR * 2));
    bf16* Rqk  = (bf16*)(ws + bump((size_t)CNR * CD * 2));
    bf16* Rv   = (bf16*)(ws + bump((size_t)CNR * CD * 2));
    bf16* WOt  = (bf16*)(ws + bump((size_t)CD * CD * 2));
    float* Hq  = (float*)(ws + bump((size_t)CBS * CR * 4));
    float* Hk  = (float*)(ws + bump((size_t)CBS * CR * 4));
    float* Hv  = (float*)(ws + bump((size_t)CBS * CR * 4));
    bf16* Qb   = (bf16*)(ws + bump((size_t)CBS * CD * 2));
    bf16* Kb   = (bf16*)(ws + bump((size_t)CBS * CD * 2));
    bf16* Vb   = (bf16*)(ws + bump((size_t)CBS * CD * 2));
    bf16* At   = (bf16*)(ws + bump((size_t)CBS * CD * 2));
    char* big  = ws + bump((size_t)CBS * CNR * 4 * 2);   // T region, reused by WH
    float* Tqk = (float*)big;
    float* Tv  = (float*)(big + (size_t)CBS * CNR * 4);
    bf16* WHq  = (bf16*)big;
    bf16* WHk  = (bf16*)(big + (size_t)CBS * CNR * 2);
    bf16* WHv  = (bf16*)(big + (size_t)CBS * CNR * 4);

    // 1. conversions
    k_cvt<<<(CBS * CD) / 256, 256, 0, stream>>>(x, Xb, CBS * CD);
    k_cvt_feat<<<(CN * CD * CR) / 256, 256, 0, stream>>>(fqk, Fqk);
    k_cvt_feat<<<(CN * CD * CR) / 256, 256, 0, stream>>>(fv, Fv);
    k_cvt<<<(CNR * CD) / 256, 256, 0, stream>>>(rqk, Rqk, CNR * CD);
    k_cvt<<<(CNR * CD) / 256, 256, 0, stream>>>(rv, Rv, CNR * CD);
    k_cvt_wot<<<(CD * CD) / 256, 256, 0, stream>>>(wo, WOt);

    // 2. feature GEMMs (f_qk shared by Q and K)
    dim3 gFeat(CNR / BN, CBS / BM);
    k_gemm<false><<<gFeat, 256, 0, stream>>>(Xb, Fqk, Tqk, CBS, CNR, CD);
    k_gemm<false><<<gFeat, 256, 0, stream>>>(Xb, Fv, Tv, CBS, CNR, CD);

    // 3. neuron-weighted reductions
    k_reduce_h<<<(CBS * CR) / 256, 256, 0, stream>>>(Tqk, wfQ, Hq);
    k_reduce_h<<<(CBS * CR) / 256, 256, 0, stream>>>(Tqk, wfK, Hk);
    k_reduce_h<<<(CBS * CR) / 256, 256, 0, stream>>>(Tv, wfV, Hv);

    // 4. restore outer products (T region reused after reductions complete)
    k_build_wh<<<(CBS * CNR) / 256, 256, 0, stream>>>(Hq, wrQ, WHq);
    k_build_wh<<<(CBS * CNR) / 256, 256, 0, stream>>>(Hk, wrK, WHk);
    k_build_wh<<<(CBS * CNR) / 256, 256, 0, stream>>>(Hv, wrV, WHv);

    // 5. restore GEMMs -> bf16 Q/K/V
    dim3 gRes(CD / BN, CBS / BM);
    k_gemm<true><<<gRes, 256, 0, stream>>>(WHq, Rqk, Qb, CBS, CD, CNR);
    k_gemm<true><<<gRes, 256, 0, stream>>>(WHk, Rqk, Kb, CBS, CD, CNR);
    k_gemm<true><<<gRes, 256, 0, stream>>>(WHv, Rv, Vb, CBS, CD, CNR);

    // 6. causal flash attention
    k_attn<<<dim3(CS / 128, CB * CH), 256, 0, stream>>>(Qb, Kb, Vb, At);

    // 7. output projection -> f32 d_out
    k_gemm<false><<<gRes, 256, 0, stream>>>(At, WOt, (float*)d_out, CBS, CD, CD);
}
